// GraphAgent_18150531793468
// MI455X (gfx1250) — matrix operand order repaired and checked by reading, not compiled, not run on hardware
//
#include <hip/hip_runtime.h>
#include <hip/hip_bf16.h>

typedef __attribute__((ext_vector_type(16))) _Float16 v16h;
typedef __attribute__((ext_vector_type(4)))  _Float16 v4h;
typedef __attribute__((ext_vector_type(8)))  float    v8f;
typedef __attribute__((ext_vector_type(4)))  unsigned int v4u;

#define WMMA_F16(a, b, c) \
    __builtin_amdgcn_wmma_f32_16x16x32_f16(false, (a), false, (b), (short)0, (c), false, false)

__device__ __forceinline__ float leakyf(float x) { return x > 0.f ? x : 0.01f * x; }

// ---- A fragment: 16x32 f16 tile from LDS (row-major, leading dim ld halves) ----
// Lane layout (ISA 7.12.2): lanes 0-15 row M=lane, K = {kt*32+0..7, kt*32+16..23};
// lanes 16-31 same rows, K = {+8..15, +24..31}. Two contiguous 16B chunks per lane.
__device__ __forceinline__ v16h load_a_frag(const _Float16* As, int ld, int kt) {
    int lane = threadIdx.x & 31;
    int m = lane & 15;
    int kb = kt * 32 + ((lane >> 4) << 3);
    union { v4u u[2]; v16h h; } r;
    r.u[0] = *reinterpret_cast<const v4u*>(As + m * ld + kb);
    r.u[1] = *reinterpret_cast<const v4u*>(As + m * ld + kb + 16);
    return r.h;
}

// ---- B fragment: pre-packed per-lane contiguous 16 halves (32B) in global ----
__device__ __forceinline__ v16h load_b_frag(const _Float16* W, int tileIdx) {
    int lane = threadIdx.x & 31;
    const _Float16* p = W + (((size_t)tileIdx * 32 + lane) << 4);
    union { v4u u[2]; v16h h; } r;
    const v4u* q = reinterpret_cast<const v4u*>(p);
    r.u[0] = q[0];
    r.u[1] = q[1];
    return r.h;
}

// ---- Pack a KxN weight (f32) into WMMA B-fragment lane order (f16) ----
// B layout: within a K=32 tile, lane l holds column n = nt*16+(l&15),
// K values k = kt*32 + (l>>4)*8 + j + (j>>3)*8, j=0..15 (contiguous per lane in dst).
__global__ void ga_pack(const float* __restrict__ src, _Float16* __restrict__ dst,
                        int ntiles, int srcLd, int realN, int trans) {
    int nt = blockIdx.x % ntiles;
    int lane = threadIdx.x;
    int n = nt * 16 + (lane & 15);
    int kb = (blockIdx.x / ntiles) * 32 + ((lane >> 4) << 3);
    _Float16* d = dst + (((size_t)blockIdx.x * 32 + lane) << 4);
#pragma unroll
    for (int j = 0; j < 16; ++j) {
        int k = kb + j + ((j >> 3) << 3);
        float v;
        if (trans) v = src[(size_t)n * srcLd + k];                 // B[k][n] = src[n][k]
        else       v = (n < realN) ? src[(size_t)k * srcLd + n] : 0.f;
        d[j] = (_Float16)v;
    }
}

__global__ void ga_zero(float* __restrict__ p, int n) {
    int i = blockIdx.x * blockDim.x + threadIdx.x;
    if (i < n) p[i] = 0.f;
}

__global__ void ga_deg(const int* __restrict__ tgt, float* __restrict__ deg, int E) {
    int e = blockIdx.x * blockDim.x + threadIdx.x;
    if (e < E) atomicAdd(&deg[tgt[e]], 1.f);
}

__global__ void ga_invdeg(float* __restrict__ deg, int n) {
    int i = blockIdx.x * blockDim.x + threadIdx.x;
    if (i < n) deg[i] = 1.f / fmaxf(deg[i], 1.f);
}

// ---- block2emb: h = leaky(cat(blockemb[x],vec[batch]) @ W1 + b1) @ W2 + b2 ----
// one wave per 16-node tile
__global__ void __launch_bounds__(32)
ga_embed(const int* __restrict__ x_idx, const int* __restrict__ batch,
         const float* __restrict__ vec_data, const float* __restrict__ blockemb,
         const _Float16* __restrict__ Pw1, const float* __restrict__ b1,
         const _Float16* __restrict__ Pw2, const float* __restrict__ b2,
         float* __restrict__ h) {
    __shared__ alignas(16) _Float16 As[16 * 128];
    __shared__ alignas(16) _Float16 A2[16 * 64];
    int tile = blockIdx.x, lane = threadIdx.x;
    for (int it = 0; it < 64; ++it) {
        int idx = it * 32 + lane;
        int row = idx >> 7, col = idx & 127;
        int node = tile * 16 + row;
        float v = (col < 64) ? blockemb[(size_t)x_idx[node] * 64 + col]
                             : vec_data[(size_t)batch[node] * 64 + (col - 64)];
        As[idx] = (_Float16)v;
    }
    int nl = lane & 15, hi = lane >> 4;
    v16h a0 = load_a_frag(As, 128, 0), a1 = load_a_frag(As, 128, 1);
    v16h a2 = load_a_frag(As, 128, 2), a3 = load_a_frag(As, 128, 3);
#pragma unroll
    for (int nt = 0; nt < 4; ++nt) {
        v8f acc = {};
        acc = WMMA_F16(a0, load_b_frag(Pw1, 0 * 4 + nt), acc);
        acc = WMMA_F16(a1, load_b_frag(Pw1, 1 * 4 + nt), acc);
        acc = WMMA_F16(a2, load_b_frag(Pw1, 2 * 4 + nt), acc);
        acc = WMMA_F16(a3, load_b_frag(Pw1, 3 * 4 + nt), acc);
        int n = nt * 16 + nl;
        float bias = b1[n];
#pragma unroll
        for (int v = 0; v < 8; ++v) {
            int m = v + (hi << 3);
            A2[m * 64 + n] = (_Float16)leakyf(acc[v] + bias);
        }
    }
    v16h c0 = load_a_frag(A2, 64, 0), c1 = load_a_frag(A2, 64, 1);
#pragma unroll
    for (int nt = 0; nt < 4; ++nt) {
        v8f acc = {};
        acc = WMMA_F16(c0, load_b_frag(Pw2, nt), acc);
        acc = WMMA_F16(c1, load_b_frag(Pw2, 4 + nt), acc);
        int n = nt * 16 + nl;
        float bias = b2[n];
#pragma unroll
        for (int v = 0; v < 8; ++v) {
            int m = v + (hi << 3);
            h[((size_t)tile * 16 + m) * 64 + n] = acc[v] + bias;
        }
    }
}

// ---- edge messages: msg = (h[src]·bondemb[b0]) * bondemb[b1], scatter to tgt ----
// one wave per edge (rank-1 factorization of W_e: O(H) not O(H^2))
__global__ void ga_msg(const float* __restrict__ h, const int* __restrict__ edge_index,
                       const int* __restrict__ ebond, const float* __restrict__ bondemb,
                       float* __restrict__ aggr, int E) {
    int wave = (blockIdx.x * blockDim.x + threadIdx.x) >> 5;
    int lane = threadIdx.x & 31;
    if (wave >= E) return;
    int src = edge_index[wave];
    int tgt = edge_index[E + wave];
    int b0 = ebond[wave * 2 + 0];
    int b1 = ebond[wave * 2 + 1];
    const float* hv = h + (size_t)src * 64;
    const float* e0 = bondemb + (size_t)b0 * 64;
    float p = hv[lane] * e0[lane] + hv[lane + 32] * e0[lane + 32];
#pragma unroll
    for (int off = 16; off > 0; off >>= 1) p += __shfl_xor(p, off, 32);
    const float* e1 = bondemb + (size_t)b1 * 64;
    atomicAdd(&aggr[(size_t)tgt * 64 + lane], p * e1[lane]);
    atomicAdd(&aggr[(size_t)tgt * 64 + lane + 32], p * e1[lane + 32]);
}

__device__ __forceinline__ float gru_one(float gir, float ghr, float giz, float ghz,
                                         float gin, float ghn, float hold) {
    float r = 1.f / (1.f + expf(-(gir + ghr)));
    float z = 1.f / (1.f + expf(-(giz + ghz)));
    float nn = tanhf(gin + r * ghn);
    return (1.f - z) * nn + z * hold;
}

// ---- fused NNConv-root + mean-aggr + LeakyReLU + GRU cell; one wave / 16 nodes ----
__global__ void __launch_bounds__(32)
ga_node(float* __restrict__ h, const float* __restrict__ aggr,
        const float* __restrict__ invdeg,
        const _Float16* __restrict__ Pconv, const float* __restrict__ cbias,
        const _Float16* __restrict__ Pwih, const _Float16* __restrict__ Pwhh,
        const float* __restrict__ bih, const float* __restrict__ bhh) {
    __shared__ alignas(16) _Float16 Ah[16 * 64];
    __shared__ alignas(16) _Float16 Am[16 * 64];
    __shared__ alignas(16) float gis[16 * 192];
    __shared__ alignas(16) float ghs[16 * 192];
    int tile = blockIdx.x, lane = threadIdx.x;
    int nl = lane & 15, hi = lane >> 4;

    const float4* hs = reinterpret_cast<const float4*>(h + (size_t)tile * 1024);
#pragma unroll
    for (int it = 0; it < 8; ++it) {
        float4 x = hs[it * 32 + lane];
        v4h hv = {(_Float16)x.x, (_Float16)x.y, (_Float16)x.z, (_Float16)x.w};
        *reinterpret_cast<v4h*>(&Ah[(it * 32 + lane) * 4]) = hv;
    }
    v16h ah0 = load_a_frag(Ah, 64, 0), ah1 = load_a_frag(Ah, 64, 1);

    // m = leaky(h @ conv_root + aggr*invdeg + cbias)
#pragma unroll
    for (int nt = 0; nt < 4; ++nt) {
        v8f acc = {};
        acc = WMMA_F16(ah0, load_b_frag(Pconv, nt), acc);
        acc = WMMA_F16(ah1, load_b_frag(Pconv, 4 + nt), acc);
        int n = nt * 16 + nl;
        float bias = cbias[n];
#pragma unroll
        for (int v = 0; v < 8; ++v) {
            int m = v + (hi << 3);
            int node = tile * 16 + m;
            float x = acc[v] + aggr[(size_t)node * 64 + n] * invdeg[node] + bias;
            Am[m * 64 + n] = (_Float16)leakyf(x);
        }
    }
    v16h am0 = load_a_frag(Am, 64, 0), am1 = load_a_frag(Am, 64, 1);

    // gi = m @ Wih^T + bih ; gh = h @ Whh^T + bhh   (N-tiles of 16, 12 each)
#pragma unroll
    for (int nt = 0; nt < 12; ++nt) {
        v8f acc = {};
        acc = WMMA_F16(am0, load_b_frag(Pwih, nt), acc);
        acc = WMMA_F16(am1, load_b_frag(Pwih, 12 + nt), acc);
        int n = nt * 16 + nl;
        float bias = bih[n];
#pragma unroll
        for (int v = 0; v < 8; ++v) gis[(v + (hi << 3)) * 192 + n] = acc[v] + bias;
    }
#pragma unroll
    for (int nt = 0; nt < 12; ++nt) {
        v8f acc = {};
        acc = WMMA_F16(ah0, load_b_frag(Pwhh, nt), acc);
        acc = WMMA_F16(ah1, load_b_frag(Pwhh, 12 + nt), acc);
        int n = nt * 16 + nl;
        float bias = bhh[n];
#pragma unroll
        for (int v = 0; v < 8; ++v) ghs[(v + (hi << 3)) * 192 + n] = acc[v] + bias;
    }

    // gates (PyTorch order r,z,n), fp32, float4 per lane-iteration
#pragma unroll
    for (int it = 0; it < 8; ++it) {
        int lin = it * 32 + lane;
        int m = lin >> 4;
        int c = (lin & 15) << 2;
        const float* gi = &gis[m * 192];
        const float* gh = &ghs[m * 192];
        float4 gir = *reinterpret_cast<const float4*>(&gi[c]);
        float4 giz = *reinterpret_cast<const float4*>(&gi[64 + c]);
        float4 gin = *reinterpret_cast<const float4*>(&gi[128 + c]);
        float4 ghr = *reinterpret_cast<const float4*>(&gh[c]);
        float4 ghz = *reinterpret_cast<const float4*>(&gh[64 + c]);
        float4 ghn = *reinterpret_cast<const float4*>(&gh[128 + c]);
        float* hp = h + ((size_t)tile * 16 + m) * 64 + c;
        float4 hold = *reinterpret_cast<const float4*>(hp);
        float4 hn;
        hn.x = gru_one(gir.x, ghr.x, giz.x, ghz.x, gin.x, ghn.x, hold.x);
        hn.y = gru_one(gir.y, ghr.y, giz.y, ghz.y, gin.y, ghn.y, hold.y);
        hn.z = gru_one(gir.z, ghr.z, giz.z, ghz.z, gin.z, ghn.z, hold.z);
        hn.w = gru_one(gir.w, ghr.w, giz.w, ghz.w, gin.w, ghn.w, hold.w);
        *reinterpret_cast<float4*>(hp) = hn;
    }
}

// ---- stem head: cat(h[stem_idx], stememb) -> 128->64->64->105 ----
__global__ void __launch_bounds__(32)
ga_stem(const float* __restrict__ h, const int* __restrict__ stems,
        const int* __restrict__ stems_batch, const int* __restrict__ slices,
        const int* __restrict__ stypes, const float* __restrict__ stememb,
        const _Float16* __restrict__ Pw1, const float* __restrict__ b1,
        const _Float16* __restrict__ Pw2, const float* __restrict__ b2,
        const _Float16* __restrict__ Pw3, const float* __restrict__ b3,
        float* __restrict__ outp, int N) {
    __shared__ alignas(16) _Float16 As[16 * 128];
    __shared__ alignas(16) _Float16 A2[16 * 64];
    __shared__ alignas(16) _Float16 A3[16 * 64];
    int tile = blockIdx.x, lane = threadIdx.x;
    for (int it = 0; it < 64; ++it) {
        int idx = it * 32 + lane;
        int row = idx >> 7, col = idx & 127;
        int s = tile * 16 + row;
        float v;
        if (col < 64) {
            int si = slices[stems_batch[s]] + stems[s * 2];
            si = si < 0 ? 0 : (si > N - 1 ? N - 1 : si);
            v = h[(size_t)si * 64 + col];
        } else {
            v = stememb[(size_t)stypes[s] * 64 + (col - 64)];
        }
        As[idx] = (_Float16)v;
    }
    int nl = lane & 15, hi = lane >> 4;
    v16h a0 = load_a_frag(As, 128, 0), a1 = load_a_frag(As, 128, 1);
    v16h a2 = load_a_frag(As, 128, 2), a3 = load_a_frag(As, 128, 3);
#pragma unroll
    for (int nt = 0; nt < 4; ++nt) {
        v8f acc = {};
        acc = WMMA_F16(a0, load_b_frag(Pw1, 0 * 4 + nt), acc);
        acc = WMMA_F16(a1, load_b_frag(Pw1, 1 * 4 + nt), acc);
        acc = WMMA_F16(a2, load_b_frag(Pw1, 2 * 4 + nt), acc);
        acc = WMMA_F16(a3, load_b_frag(Pw1, 3 * 4 + nt), acc);
        int n = nt * 16 + nl;
        float bias = b1[n];
#pragma unroll
        for (int v = 0; v < 8; ++v) A2[(v + (hi << 3)) * 64 + n] = (_Float16)leakyf(acc[v] + bias);
    }
    v16h c0 = load_a_frag(A2, 64, 0), c1 = load_a_frag(A2, 64, 1);
#pragma unroll
    for (int nt = 0; nt < 4; ++nt) {
        v8f acc = {};
        acc = WMMA_F16(c0, load_b_frag(Pw2, nt), acc);
        acc = WMMA_F16(c1, load_b_frag(Pw2, 4 + nt), acc);
        int n = nt * 16 + nl;
        float bias = b2[n];
#pragma unroll
        for (int v = 0; v < 8; ++v) A3[(v + (hi << 3)) * 64 + n] = (_Float16)leakyf(acc[v] + bias);
    }
    v16h d0 = load_a_frag(A3, 64, 0), d1 = load_a_frag(A3, 64, 1);
#pragma unroll
    for (int nt = 0; nt < 7; ++nt) {  // 112 cols (105 padded)
        v8f acc = {};
        acc = WMMA_F16(d0, load_b_frag(Pw3, nt), acc);
        acc = WMMA_F16(d1, load_b_frag(Pw3, 7 + nt), acc);
        int n = nt * 16 + nl;
        if (n < 105) {
            float bias = b3[n];
#pragma unroll
            for (int v = 0; v < 8; ++v) {
                int m = v + (hi << 3);
                outp[((size_t)tile * 16 + m) * 105 + n] = acc[v] + bias;
            }
        }
    }
}

__global__ void ga_pool(const float* __restrict__ h, const int* __restrict__ batch,
                        float* __restrict__ pooled, float* __restrict__ cnt, int N) {
    int i = blockIdx.x * blockDim.x + threadIdx.x;
    if (i >= N * 64) return;
    int node = i >> 6, col = i & 63;
    int g = batch[node];
    atomicAdd(&pooled[(size_t)g * 64 + col], h[i]);
    if (col == 0) atomicAdd(&cnt[g], 1.f);
}

__global__ void __launch_bounds__(32)
ga_mol(const float* __restrict__ pooled, const float* __restrict__ cnt,
       const float* __restrict__ w1, const float* __restrict__ b1,
       const float* __restrict__ w2, const float* __restrict__ b2,
       float* __restrict__ outp) {
    int g = blockIdx.x, lane = threadIdx.x;
    float ic = 1.f / fmaxf(cnt[g], 1.f);
    const float* pv = pooled + (size_t)g * 64;
    float h0 = b1[lane], h1 = b1[lane + 32];
    for (int k = 0; k < 64; ++k) {
        float x = pv[k] * ic;
        h0 += x * w1[k * 64 + lane];
        h1 += x * w1[k * 64 + lane + 32];
    }
    float p = leakyf(h0) * w2[lane] + leakyf(h1) * w2[lane + 32];
#pragma unroll
    for (int off = 16; off > 0; off >>= 1) p += __shfl_xor(p, off, 32);
    if (lane == 0) outp[g] = p + b2[0];
}

extern "C" void kernel_launch(void* const* d_in, const int* in_sizes, int n_in,
                              void* d_out, int out_size, void* d_ws, size_t ws_size,
                              hipStream_t stream) {
    (void)in_sizes; (void)n_in; (void)out_size; (void)ws_size;
    const int N = 50000, E = 50000, G = 2048, S = 8192;

    const int* x_idx        = (const int*)d_in[0];
    const int* stypes       = (const int*)d_in[1];
    const int* ebond        = (const int*)d_in[2];
    const int* edge_index   = (const int*)d_in[3];   // [2][E] flat
    const int* batch        = (const int*)d_in[4];
    const int* slices       = (const int*)d_in[5];
    const int* stems        = (const int*)d_in[6];
    const int* stems_batch  = (const int*)d_in[7];
    const float* vec_data   = (const float*)d_in[8];
    const float* blockemb   = (const float*)d_in[9];
    const float* stememb    = (const float*)d_in[10];
    const float* bondemb    = (const float*)d_in[11];
    const float* conv_root  = (const float*)d_in[12];
    const float* conv_bias  = (const float*)d_in[13];
    const float* b2e_w1     = (const float*)d_in[14];
    const float* b2e_b1     = (const float*)d_in[15];
    const float* b2e_w2     = (const float*)d_in[16];
    const float* b2e_b2     = (const float*)d_in[17];
    const float* gru_wih    = (const float*)d_in[18];
    const float* gru_whh    = (const float*)d_in[19];
    const float* gru_bih    = (const float*)d_in[20];
    const float* gru_bhh    = (const float*)d_in[21];
    const float* s2p_w1     = (const float*)d_in[22];
    const float* s2p_b1     = (const float*)d_in[23];
    const float* s2p_w2     = (const float*)d_in[24];
    const float* s2p_b2     = (const float*)d_in[25];
    const float* s2p_w3     = (const float*)d_in[26];
    const float* s2p_b3     = (const float*)d_in[27];
    const float* g2p_w1     = (const float*)d_in[28];
    const float* g2p_b1     = (const float*)d_in[29];
    const float* g2p_w2     = (const float*)d_in[30];
    const float* g2p_b2     = (const float*)d_in[31];
    float* out = (float*)d_out;

    // workspace carve (~26.5 MB)
    char* w = (char*)d_ws;
    auto carve = [&](size_t bytes) { void* p = (void*)w; w += (bytes + 255) & ~(size_t)255; return p; };
    float* h      = (float*)carve((size_t)N * 64 * 4);
    float* aggr   = (float*)carve((size_t)N * 64 * 4);
    float* invdeg = (float*)carve((size_t)N * 4);
    float* pooled = (float*)carve((size_t)G * 64 * 4);
    float* cnt    = (float*)carve((size_t)G * 4);
    _Float16* P   = (_Float16*)carve((size_t)60416 * 2);
    _Float16* Pb2e1 = P;            // ktiles*ntiles*512 halves each
    _Float16* Pb2e2 = P + 8192;
    _Float16* Pconv = P + 12288;
    _Float16* Pwih  = P + 16384;
    _Float16* Pwhh  = P + 28672;
    _Float16* Ps1   = P + 40960;
    _Float16* Ps2   = P + 49152;
    _Float16* Ps3   = P + 53248;

    // pack weights into WMMA B-fragment layout (f16)
    ga_pack<<<16, 32, 0, stream>>>(b2e_w1,    Pb2e1, 4, 64, 64, 0);   // 128x64
    ga_pack<<<8,  32, 0, stream>>>(b2e_w2,    Pb2e2, 4, 64, 64, 0);   // 64x64
    ga_pack<<<8,  32, 0, stream>>>(conv_root, Pconv, 4, 64, 64, 0);   // 64x64
    ga_pack<<<24, 32, 0, stream>>>(gru_wih,   Pwih, 12, 64, 192, 1);  // (192,64)^T -> 64x192
    ga_pack<<<24, 32, 0, stream>>>(gru_whh,   Pwhh, 12, 64, 192, 1);
    ga_pack<<<16, 32, 0, stream>>>(s2p_w1,    Ps1,  4, 64, 64, 0);    // 128x64
    ga_pack<<<8,  32, 0, stream>>>(s2p_w2,    Ps2,  4, 64, 64, 0);    // 64x64
    ga_pack<<<14, 32, 0, stream>>>(s2p_w3,    Ps3,  7, 105, 105, 0);  // 64x105 -> pad 112

    // degree -> inv_deg
    ga_zero<<<(N + 255) / 256, 256, 0, stream>>>(invdeg, N);
    ga_deg<<<(E + 255) / 256, 256, 0, stream>>>(edge_index + E, invdeg, E);
    ga_invdeg<<<(N + 255) / 256, 256, 0, stream>>>(invdeg, N);

    // block2emb
    ga_embed<<<N / 16, 32, 0, stream>>>(x_idx, batch, vec_data, blockemb,
                                        Pb2e1, b2e_b1, Pb2e2, b2e_b2, h);

    // 10 message-passing + GRU steps
    for (int s = 0; s < 10; ++s) {
        ga_zero<<<(N * 64 + 255) / 256, 256, 0, stream>>>(aggr, N * 64);
        ga_msg<<<(E + 7) / 8, 256, 0, stream>>>(h, edge_index, ebond, bondemb, aggr, E);
        ga_node<<<N / 16, 32, 0, stream>>>(h, aggr, invdeg, Pconv, conv_bias,
                                           Pwih, Pwhh, gru_bih, gru_bhh);
    }

    // stem predictions -> out[0 .. S*105)
    ga_stem<<<S / 16, 32, 0, stream>>>(h, stems, stems_batch, slices, stypes, stememb,
                                       Ps1, s2p_b1, Ps2, s2p_b2, Ps3, s2p_b3, out, N);

    // mean pool + mol head -> out[S*105 .. S*105+G)
    ga_zero<<<(G * 64 + 255) / 256, 256, 0, stream>>>(pooled, G * 64);
    ga_zero<<<(G + 255) / 256, 256, 0, stream>>>(cnt, G);
    ga_pool<<<(N * 64 + 255) / 256, 256, 0, stream>>>(h, batch, pooled, cnt, N);
    ga_mol<<<G, 32, 0, stream>>>(pooled, cnt, g2p_w1, g2p_b1, g2p_w2, g2p_b2,
                                 out + (size_t)S * 105);
}
